// VAE_decoder_40621800686258
// MI455X (gfx1250) — compile-verified, matrix operand order checked by reading, not run on hardware
//
#include <hip/hip_runtime.h>

// ---------------- problem constants ----------------
#define BATCH   4096
#define MOTW    1792          // MOTION_SIZE*(USED_MOTIONS-1)
#define ANGW    448           // USED_ANGLES*(USED_MOTIONS-1)
#define LAT     128
#define IN_SZ   2368          // MOTW+ANGW+LAT
#define H1SZ    2048
#define H2SZ    2368
#define NEXP    8
#define K3SZ    (IN_SZ + LAT) // 2496
#define K4SZ    (H1SZ + LAT)  // 2176
#define K5SZ    (H2SZ + LAT)  // 2496
#define OUTN    256

// ---------------- GEMM tiling ----------------
#define BM   256
#define BN   64
#define BK   32
#define LDSTR (BK + 8)        // padded LDS row (elements) to dodge bank conflicts

typedef __bf16 bf16;
typedef __attribute__((ext_vector_type(16))) __bf16 v16bf;
typedef __attribute__((ext_vector_type(8)))  __bf16 v8bf;
typedef __attribute__((ext_vector_type(8)))  float  v8f;

// exact pointee type the async-LDS builtins expect: gcc-style vector of 4 ints
typedef int v4i __attribute__((vector_size(16)));
typedef __attribute__((address_space(1))) v4i* gptr_v4i;  // global
typedef __attribute__((address_space(3))) v4i* lptr_v4i;  // LDS

static __device__ __forceinline__ float elu(float v) {
    return v > 0.f ? v : (__expf(v) - 1.f);
}

// 16B global -> LDS copy. Preferred path: CDNA5 async direct-to-LDS (ASYNCcnt),
// no VGPR staging. Fallback: staged copy through registers.
static __device__ __forceinline__ void async_cp16(const bf16* g, bf16* l) {
#if __has_builtin(__builtin_amdgcn_global_load_async_to_lds_b128)
    __builtin_amdgcn_global_load_async_to_lds_b128(
        (gptr_v4i)(unsigned long long)g,
        (lptr_v4i)(unsigned)(unsigned long long)l,
        0, 0);
#else
    *(uint4*)l = *(const uint4*)g;
#endif
}

static __device__ __forceinline__ void wait_async() {
#if __has_builtin(__builtin_amdgcn_s_wait_asynccnt)
    __builtin_amdgcn_s_wait_asynccnt(0);
#else
    asm volatile("s_wait_asynccnt 0x0" ::: "memory");
#endif
}

// Issue the 5 async 16B copies for one (A,B) K-chunk of the block tile.
static __device__ __forceinline__ void load_tile(
    const bf16* __restrict__ A, int lda, int mBase,
    const bf16* __restrict__ W, int ldw, int nBase,
    int k0, bf16 (*As)[LDSTR], bf16 (*Bs)[LDSTR], int lrow, int lcol)
{
    const bf16* ap = A + (size_t)(mBase + lrow) * lda + k0 + lcol;
#pragma unroll
    for (int p = 0; p < 4; ++p)
        async_cp16(ap + (size_t)(p * 64) * lda, &As[lrow + p * 64][lcol]);
    async_cp16(W + (size_t)(nBase + lrow) * ldw + k0 + lcol, &Bs[lrow][lcol]);
}

// A/B fragment for v_wmma_f32_16x16x32_bf16 (ISA 7.12.2, 16-bit 16x32):
// lane(0..15): row M=lane, K = {0..7, 16..23}; lane(16..31): row M=lane-16, K = {8..15, 24..31}
static __device__ __forceinline__ v16bf load_frag(const bf16* row, int half) {
    v8bf lo = *(const v8bf*)(row + half * 8);
    v8bf hi = *(const v8bf*)(row + 16 + half * 8);
    v16bf f;
#pragma unroll
    for (int i = 0; i < 8; ++i) { f[i] = lo[i]; f[i + 8] = hi[i]; }
    return f;
}

// ================= generic bf16 GEMM + bias + ELU =================
// C[m,n] = elu( sum_k A[m,k]*W[n,k] + bias[n] ), A:[M,lda] bf16, W:[N,K] bf16 row-major.
// blockIdx.z selects expert via {wStride,bStride,cStride}.
template <typename OutT>
__global__ __launch_bounds__(256) void gemm_bf16_elu(
    const bf16* __restrict__ A, int lda,
    const bf16* __restrict__ W, long wStride,
    const float* __restrict__ bias, int bStride,
    OutT* __restrict__ C, long cStride, int ldc,
    int K)
{
    __shared__ bf16 As[2][BM][LDSTR];
    __shared__ bf16 Bs[2][BN][LDSTR];

    const int tid   = threadIdx.x;
    const int nBase = blockIdx.x * BN;
    const int mBase = blockIdx.y * BM;
    W    += (long)blockIdx.z * wStride;
    bias += (long)blockIdx.z * bStride;
    C    += (long)blockIdx.z * cStride;

    const int lrow = tid >> 2;        // 0..63
    const int lcol = (tid & 3) * 8;   // 0,8,16,24

    const int lane = tid & 31;
    const int wave = tid >> 5;
    const int wm   = wave & 3;        // 4 waves along M
    const int wn   = wave >> 2;       // 2 waves along N
    const int half = lane >> 4;
    const int l16  = lane & 15;

    v8f acc[4][2] = {};
    const int nkt = K / BK;

    load_tile(A, lda, mBase, W, K, nBase, 0, As[0], Bs[0], lrow, lcol);
    wait_async();
    __syncthreads();

    for (int kt = 0; kt < nkt; ++kt) {
        const int buf = kt & 1;
        if (kt + 1 < nkt)   // async-fill the other buffer while we compute
            load_tile(A, lda, mBase, W, K, nBase, (kt + 1) * BK,
                      As[buf ^ 1], Bs[buf ^ 1], lrow, lcol);

        v16bf af[4], bfv[2];
#pragma unroll
        for (int mi = 0; mi < 4; ++mi)
            af[mi] = load_frag(&As[buf][wm * 64 + mi * 16 + l16][0], half);
#pragma unroll
        for (int ni = 0; ni < 2; ++ni)
            bfv[ni] = load_frag(&Bs[buf][wn * 32 + ni * 16 + l16][0], half);
#pragma unroll
        for (int mi = 0; mi < 4; ++mi)
#pragma unroll
            for (int ni = 0; ni < 2; ++ni)
                acc[mi][ni] = __builtin_amdgcn_wmma_f32_16x16x32_bf16(
                    false, af[mi], false, bfv[ni], (short)0, acc[mi][ni], false, false);

        if (kt + 1 < nkt) wait_async();
        __syncthreads();
    }

    // epilogue: C/D layout — VGPR r: lanes0-15 -> M=r, lanes16-31 -> M=r+8
#pragma unroll
    for (int mi = 0; mi < 4; ++mi)
#pragma unroll
        for (int ni = 0; ni < 2; ++ni) {
            const int gm = mBase + wm * 64 + mi * 16 + half * 8;
            const int gn = nBase + wn * 32 + ni * 16 + l16;
            const float bv = bias[gn];
#pragma unroll
            for (int r = 0; r < 8; ++r) {
                const float v = elu(acc[mi][ni][r] + bv);
                C[(size_t)(gm + r) * ldc + gn] = (OutT)v;
            }
        }
}

// ================= MoE layer 2: out[b,k] = sum_e para[e]*elu(h1[e]·w2[e]^T + b2[e]) =================
__global__ __launch_bounds__(256) void moe_layer2(
    const bf16* __restrict__ H1b,   // [NEXP][BATCH][H1SZ]
    const bf16* __restrict__ W2,    // [NEXP][H2SZ][H1SZ]
    const float* __restrict__ B2,   // [NEXP][H2SZ]
    const float* __restrict__ para, // [NEXP]
    bf16* __restrict__ C, int ldc)  // [BATCH][K3SZ], cols [0,H2SZ)
{
    __shared__ bf16 As[2][BM][LDSTR];
    __shared__ bf16 Bs[2][BN][LDSTR];

    const int tid   = threadIdx.x;
    const int nBase = blockIdx.x * BN;
    const int mBase = blockIdx.y * BM;

    const int lrow = tid >> 2;
    const int lcol = (tid & 3) * 8;
    const int lane = tid & 31;
    const int wave = tid >> 5;
    const int wm   = wave & 3;
    const int wn   = wave >> 2;
    const int half = lane >> 4;
    const int l16  = lane & 15;

    v8f facc[4][2] = {};
    const int nkt = H1SZ / BK;   // 64

    for (int e = 0; e < NEXP; ++e) {
        const bf16* A = H1b + (size_t)e * BATCH * H1SZ;
        const bf16* W = W2  + (size_t)e * H2SZ * H1SZ;
        const float pv = para[e];            // uniform scalar load, no movrel
        v8f acc[4][2] = {};

        load_tile(A, H1SZ, mBase, W, H1SZ, nBase, 0, As[0], Bs[0], lrow, lcol);
        wait_async();
        __syncthreads();

        for (int kt = 0; kt < nkt; ++kt) {
            const int buf = kt & 1;
            if (kt + 1 < nkt)
                load_tile(A, H1SZ, mBase, W, H1SZ, nBase, (kt + 1) * BK,
                          As[buf ^ 1], Bs[buf ^ 1], lrow, lcol);

            v16bf af[4], bfv[2];
#pragma unroll
            for (int mi = 0; mi < 4; ++mi)
                af[mi] = load_frag(&As[buf][wm * 64 + mi * 16 + l16][0], half);
#pragma unroll
            for (int ni = 0; ni < 2; ++ni)
                bfv[ni] = load_frag(&Bs[buf][wn * 32 + ni * 16 + l16][0], half);
#pragma unroll
            for (int mi = 0; mi < 4; ++mi)
#pragma unroll
                for (int ni = 0; ni < 2; ++ni)
                    acc[mi][ni] = __builtin_amdgcn_wmma_f32_16x16x32_bf16(
                        false, af[mi], false, bfv[ni], (short)0, acc[mi][ni], false, false);

            if (kt + 1 < nkt) wait_async();
            __syncthreads();   // also makes LDS safe for next expert's prologue
        }

        // fold this expert into the final accumulator
#pragma unroll
        for (int mi = 0; mi < 4; ++mi)
#pragma unroll
            for (int ni = 0; ni < 2; ++ni) {
                const int gn = nBase + wn * 32 + ni * 16 + l16;
                const float bv = B2[e * H2SZ + gn];
#pragma unroll
                for (int r = 0; r < 8; ++r)
                    facc[mi][ni][r] += pv * elu(acc[mi][ni][r] + bv);
            }
    }

#pragma unroll
    for (int mi = 0; mi < 4; ++mi)
#pragma unroll
        for (int ni = 0; ni < 2; ++ni) {
            const int gm = mBase + wm * 64 + mi * 16 + half * 8;
            const int gn = nBase + wn * 32 + ni * 16 + l16;
#pragma unroll
            for (int r = 0; r < 8; ++r)
                C[(size_t)(gm + r) * ldc + gn] = (bf16)facc[mi][ni][r];
        }
}

// ================= input concat + z-tails (fp32 -> bf16) =================
__global__ void pack_inputs(const float* __restrict__ motions,
                            const float* __restrict__ angles,
                            const float* __restrict__ z,
                            bf16* __restrict__ xb,   // [BATCH][IN_SZ]
                            bf16* __restrict__ a3,   // [BATCH][K3SZ] tail cols get z
                            bf16* __restrict__ a4)   // [BATCH][K4SZ] tail cols get z
{
    const int b = blockIdx.x;
    for (int c = threadIdx.x; c < IN_SZ; c += blockDim.x) {
        float v;
        if (c < MOTW)             v = motions[(size_t)b * MOTW + c];
        else if (c < MOTW + ANGW) v = angles[(size_t)b * ANGW + (c - MOTW)];
        else                      v = z[(size_t)b * LAT + (c - MOTW - ANGW)];
        xb[(size_t)b * IN_SZ + c] = (bf16)v;
    }
    if (threadIdx.x < LAT) {
        const float zv = z[(size_t)b * LAT + threadIdx.x];
        a3[(size_t)b * K3SZ + IN_SZ + threadIdx.x] = (bf16)zv;
        a4[(size_t)b * K4SZ + H1SZ + threadIdx.x]  = (bf16)zv;
    }
}

// ================= weight conversion fp32 -> bf16 (4 elems/thread) =================
__global__ void cvt_f32_bf16(const float* __restrict__ src, bf16* __restrict__ dst, int n4)
{
    int i = blockIdx.x * blockDim.x + threadIdx.x;
    const int stride = gridDim.x * blockDim.x;
    for (; i < n4; i += stride) {
        const float4 v = ((const float4*)src)[i];
        union { bf16 h[4]; uint2 u; } pk;
        pk.h[0] = (bf16)v.x; pk.h[1] = (bf16)v.y; pk.h[2] = (bf16)v.z; pk.h[3] = (bf16)v.w;
        ((uint2*)dst)[i] = pk.u;
    }
}

// ================= host launcher =================
extern "C" void kernel_launch(void* const* d_in, const int* in_sizes, int n_in,
                              void* d_out, int out_size, void* d_ws, size_t ws_size,
                              hipStream_t stream)
{
    const float* motions = (const float*)d_in[0];
    const float* angles  = (const float*)d_in[1];
    const float* z       = (const float*)d_in[2];
    const float* l1_w    = (const float*)d_in[3];
    const float* l1_b    = (const float*)d_in[4];
    const float* l2_w    = (const float*)d_in[5];
    const float* l2_b    = (const float*)d_in[6];
    const float* para    = (const float*)d_in[7];
    const float* l3_w    = (const float*)d_in[8];
    const float* l3_b    = (const float*)d_in[9];
    const float* l4_w    = (const float*)d_in[10];
    const float* l4_b    = (const float*)d_in[11];
    const float* l5_w    = (const float*)d_in[12];
    const float* l5_b    = (const float*)d_in[13];
    float* out = (float*)d_out;

    char* ws = (char*)d_ws;
    auto carve = [&](size_t elems) {
        bf16* p = (bf16*)ws;
        ws += (elems * sizeof(bf16) + 255) & ~(size_t)255;
        return p;
    };
    bf16* xb  = carve((size_t)BATCH * IN_SZ);        // 19.4 MB
    bf16* h1  = carve((size_t)NEXP * BATCH * H1SZ);  // 134 MB
    bf16* a3  = carve((size_t)BATCH * K3SZ);         // 20.4 MB (layer2 out; reused as layer4 out)
    bf16* a4  = carve((size_t)BATCH * K4SZ);         // 17.8 MB
    bf16* l1b = carve((size_t)NEXP * H1SZ * IN_SZ);  // 77.6 MB
    bf16* l2b = carve((size_t)NEXP * H2SZ * H1SZ);   // 77.6 MB
    bf16* l3b = carve((size_t)H1SZ * K3SZ);
    bf16* l4b = carve((size_t)H2SZ * K4SZ);
    bf16* l5b = carve((size_t)OUTN * K5SZ);

    // weight conversions (redone every call -> deterministic, capture-safe)
    cvt_f32_bf16<<<4096, 256, 0, stream>>>(l1_w, l1b, NEXP * H1SZ * IN_SZ / 4);
    cvt_f32_bf16<<<4096, 256, 0, stream>>>(l2_w, l2b, NEXP * H2SZ * H1SZ / 4);
    cvt_f32_bf16<<<1024, 256, 0, stream>>>(l3_w, l3b, H1SZ * K3SZ / 4);
    cvt_f32_bf16<<<1024, 256, 0, stream>>>(l4_w, l4b, H2SZ * K4SZ / 4);
    cvt_f32_bf16<<<256,  256, 0, stream>>>(l5_w, l5b, OUTN * K5SZ / 4);
    pack_inputs<<<BATCH, 256, 0, stream>>>(motions, angles, z, xb, a3, a4);

    // L1: x[4096,2368] x l1_w[e][2048,2368]^T -> h1[e][4096,2048], elu
    gemm_bf16_elu<bf16><<<dim3(H1SZ / BN, BATCH / BM, NEXP), 256, 0, stream>>>(
        xb, IN_SZ, l1b, (long)H1SZ * IN_SZ, l1_b, H1SZ,
        h1, (long)BATCH * H1SZ, H1SZ, IN_SZ);

    // L2 (MoE reduce): -> a3 cols [0,2368)
    moe_layer2<<<dim3(H2SZ / BN, BATCH / BM), 256, 0, stream>>>(
        h1, l2b, l2_b, para, a3, K3SZ);

    // L3: a3[4096,2496] x l3_w[2048,2496]^T -> a4 cols [0,2048), elu
    gemm_bf16_elu<bf16><<<dim3(H1SZ / BN, BATCH / BM, 1), 256, 0, stream>>>(
        a3, K3SZ, l3b, 0, l3_b, 0, a4, 0, K4SZ, K3SZ);

    // L4: a4[4096,2176] x l4_w[2368,2176]^T -> a3 cols [0,2368), elu (z-tail intact)
    gemm_bf16_elu<bf16><<<dim3(H2SZ / BN, BATCH / BM, 1), 256, 0, stream>>>(
        a4, K4SZ, l4b, 0, l4_b, 0, a3, 0, K3SZ, K4SZ);

    // L5: a3[4096,2496] x l5_w[256,2496]^T -> out f32 [4096,256], elu
    gemm_bf16_elu<float><<<dim3(OUTN / BN, BATCH / BM, 1), 256, 0, stream>>>(
        a3, K3SZ, l5b, 0, l5_b, 0, out, 0, OUTN, K3SZ);

    (void)in_sizes; (void)n_in; (void)out_size; (void)ws_size;
}
